// PrimaryLayer_33526514712962
// MI455X (gfx1250) — matrix operand order repaired and checked by reading, not compiled, not run on hardware
//
#include <hip/hip_runtime.h>
#include <hip/hip_bf16.h>

// CDNA5 / gfx1250 vector types
typedef __attribute__((ext_vector_type(16))) __bf16 v16bf;
typedef __attribute__((ext_vector_type(8)))  float  v8f;
typedef __attribute__((ext_vector_type(4)))  unsigned int u32x4;

// ---------------------------------------------------------------------------
// fp32 -> bf16 (round-to-nearest-even)
// ---------------------------------------------------------------------------
__device__ __forceinline__ unsigned short f32_to_bf16(float f) {
  unsigned int u = __builtin_bit_cast(unsigned int, f);
  u += 0x7fffu + ((u >> 16) & 1u);
  return (unsigned short)(u >> 16);
}

// ---------------------------------------------------------------------------
// Kernel 1: pack weights W[o=32][c=256][kh=9][kw=9] (fp32) into bf16
// A-fragment layout [kpos(81)][cb(8)][mtile(2)][lane(32)][i(16)], matching the
// CDNA5 16-bit A 16x32 WMMA layout:
//   lane<16 : i=0..7 -> K=i,    i=8..15 -> K=i+8   (K 0..7, 16..23)
//   lane>=16: i=0..7 -> K=i+8,  i=8..15 -> K=i+16  (K 8..15, 24..31)
// ---------------------------------------------------------------------------
__global__ void pack_w_kernel(const float* __restrict__ W,
                              unsigned short* __restrict__ Apack) {
  int t = blockIdx.x * blockDim.x + threadIdx.x;   // 81*8*2*32 = 41472 threads
  if (t >= 81 * 8 * 2 * 32) return;
  const int lane  = t & 31;
  const int mtile = (t >> 5) & 1;
  const int cb    = (t >> 6) & 7;
  const int kpos  = t >> 9;                        // kh*9+kw
  const int o     = mtile * 16 + (lane & 15);
  const int hi    = lane >> 4;
  unsigned short* dst = Apack + (size_t)t * 16;
#pragma unroll
  for (int i = 0; i < 16; ++i) {
    const int k = (i < 8) ? (hi ? i + 8 : i) : (hi ? i + 16 : i + 8);
    const int c = cb * 32 + k;
    dst[i] = f32_to_bf16(W[((size_t)o * 256 + c) * 81 + kpos]);
  }
}

// ---------------------------------------------------------------------------
// Kernel 2: fused implicit-GEMM conv + squash.
//   Block  = one batch x 4 output rows (112 positions = 7 N-tiles of 16)
//   Waves  = 7; each wave owns a 32x16 output block (2 acc tiles of v8f).
//   K loop = 8 channel chunks (cb) x 81 kernel positions.
//   Per cb: stage 15x64x32 input tile fp32->bf16 into LDS ([pixel][channel])
//           with float4-vectorized global reads, then feed B fragments with
//           ds_load_b128 (imm offsets) and A fragments with global_load_b128
//           from the packed, L2-resident weight stream.
// ---------------------------------------------------------------------------
#define XTILE_ELEMS (15 * 64 * 32)   // 30720 bf16 = 60 KB LDS
#define XTILE_VEC   (XTILE_ELEMS / 4)

__global__ __launch_bounds__(224)
void conv_squash_kernel(const unsigned short* __restrict__ Apack,
                        const float* __restrict__ X,
                        const float* __restrict__ bias,
                        float* __restrict__ out) {
  __shared__ unsigned short xtile[XTILE_ELEMS];

  const int tid  = threadIdx.x;
  const int wave = tid >> 5;           // 0..6  -> N-tile id
  const int lane = tid & 31;
  const int lo   = lane & 15;
  const int hi   = lane >> 4;
  const int b    = blockIdx.y;         // batch
  const int grp  = blockIdx.x;         // 0..6 : output-row group (4 rows)
  const int ih0  = grp * 8;            // first input row of the 15-row tile

  // This wave's 16 output positions (all valid: 112 positions tile 4 full rows)
  const int n_local = wave * 16 + lo;          // 0..111 within group
  const int loh = n_local / 28;                // local output row 0..3
  const int low = n_local % 28;                // output col
  const int base_pix = (2 * loh) * 64 + 2 * low;
  const unsigned lds_base = (unsigned)(base_pix * 64 + hi * 16); // bytes

  v8f acc0 = (v8f){0.f, 0.f, 0.f, 0.f, 0.f, 0.f, 0.f, 0.f};
  v8f acc1 = acc0;

  union Frag { v16bf v; u32x4 q[2]; };

  const float* xb = X + ((size_t)b * 256) * 4096;   // batch base (NCHW)

  for (int cb = 0; cb < 8; ++cb) {
    __syncthreads();   // protect LDS tile from previous iteration's readers

    // ---- stage: fp32 NCHW -> bf16 LDS[pixel][channel].
    // float4 along iw (rows are 64 floats, 16B aligned): 1 global_load_b128
    // + 4 converts + 4 ds_store_b16 per 4 elements.
    for (int i = tid; i < XTILE_VEC; i += 224) {
      const int pv = i % 240;                  // vec-pixel: r*16 + iw/4
      const int cc = i / 240;                  // 0..31
      const int r  = pv >> 4;                  // 0..14
      const int iw = (pv & 15) << 2;           // 0,4,...,60
      const float4 f4 = *(const float4*)(
          xb + ((size_t)(cb * 32 + cc) * 64 + (ih0 + r)) * 64 + iw);
      const int pbase = (r * 64 + iw) * 32 + cc;
      xtile[pbase]      = f32_to_bf16(f4.x);
      xtile[pbase + 32] = f32_to_bf16(f4.y);
      xtile[pbase + 64] = f32_to_bf16(f4.z);
      xtile[pbase + 96] = f32_to_bf16(f4.w);
    }
    __syncthreads();

    // ---- compute: 81 kernel positions against this 32-channel chunk
    const unsigned short* aptr =
        Apack + (size_t)cb * 1024 + (size_t)lane * 16;
#pragma unroll 1
    for (int kh = 0; kh < 9; ++kh) {
#pragma unroll
      for (int kw = 0; kw < 9; ++kw) {
        const int kpos = kh * 9 + kw;
        const unsigned short* abase = aptr + (size_t)kpos * 8192;
        if (kpos < 80)   // WGP-scope speculative prefetch of next A tile
          __builtin_prefetch(abase + 8192, 0, 3);
        Frag a0, a1, bf;
        a0.q[0] = *(const u32x4*)(abase);
        a0.q[1] = *(const u32x4*)(abase + 8);
        a1.q[0] = *(const u32x4*)(abase + 512);
        a1.q[1] = *(const u32x4*)(abase + 520);
        const u32x4* bp = (const u32x4*)((const char*)xtile + lds_base
                                         + (unsigned)((kh * 64 + kw) * 64));
        bf.q[0] = bp[0];
        bf.q[1] = bp[2];
        acc0 = __builtin_amdgcn_wmma_f32_16x16x32_bf16(
            false, a0.v, false, bf.v, (short)0, acc0, false, false);
        acc1 = __builtin_amdgcn_wmma_f32_16x16x32_bf16(
            false, a1.v, false, bf.v, (short)0, acc1, false, false);
      }
    }
  }

  // ---- epilogue: bias + closed-form squash (T=8 identical copies), x8 store.
  // C/D layout: lane0-15 = N, vgpr r -> M = r (lanes 0-15) / r+8 (lanes 16-31)
  const int n_pos = grp * 4 * 28 + n_local;      // position within 28x28 image
  const size_t out_b = (size_t)b * 25088;
#pragma unroll
  for (int mt = 0; mt < 2; ++mt) {
    const v8f a = mt ? acc1 : acc0;
#pragma unroll
    for (int r = 0; r < 8; ++r) {
      const int o = mt * 16 + r + hi * 8;
      const float v  = a[r] + bias[o];
      const float sq = 8.0f * v * v;
      const float scale = (sq / (1.0f + sq)) * rsqrtf(sq + 1e-8f);
      const float sv = v * scale;
      const size_t nout = (size_t)o * 784 + (size_t)n_pos;
      float4* dst = (float4*)(out + (out_b + nout) * 8);
      const float4 f4 = make_float4(sv, sv, sv, sv);
      dst[0] = f4;
      dst[1] = f4;
    }
  }
}

// ---------------------------------------------------------------------------
// Launch: pack_w -> fused conv (serialized on stream).
// Workspace: Apack = 81*8*1024 bf16 = 1.33 MB.
// ---------------------------------------------------------------------------
extern "C" void kernel_launch(void* const* d_in, const int* in_sizes, int n_in,
                              void* d_out, int out_size, void* d_ws, size_t ws_size,
                              hipStream_t stream) {
  (void)in_sizes; (void)n_in; (void)out_size; (void)ws_size;
  const float* x    = (const float*)d_in[0];   // [32,256,64,64]
  const float* W    = (const float*)d_in[1];   // [32,256,9,9]
  const float* bias = (const float*)d_in[2];   // [32]
  float* out = (float*)d_out;                  // [32,25088,8,1]

  unsigned short* Apack = (unsigned short*)d_ws;

  pack_w_kernel<<<(41472 + 127) / 128, 128, 0, stream>>>(W, Apack);
  conv_squash_kernel<<<dim3(7, 32), 224, 0, stream>>>(Apack, x, bias, out);
}
